// MultiHeadAttention_33784212750785
// MI455X (gfx1250) — matrix operand order repaired and checked by reading, not compiled, not run on hardware
//
#include <hip/hip_runtime.h>
#include <hip/hip_bf16.h>
#include <math.h>

typedef __attribute__((ext_vector_type(16))) _Float16 v16h;
typedef __attribute__((ext_vector_type(8)))  _Float16 v8h;
typedef __attribute__((ext_vector_type(8)))  float    v8f;

#define D_MODEL  1024
#define N_HEADS  16
#define HEAD_DIM 64

// Concatenate two aligned 16B chunks into one WMMA operand (two ds/global b128 loads).
__device__ __forceinline__ v16h cat8(const _Float16* lo, const _Float16* hi) {
  v8h a = *(const v8h*)lo;
  v8h b = *(const v8h*)hi;
  return __builtin_shufflevector(a, b, 0, 1, 2, 3, 4, 5, 6, 7, 8, 9, 10, 11, 12, 13, 14, 15);
}

__global__ void cvt_f32_f16(const float* __restrict__ in, _Float16* __restrict__ out, int n) {
  int i = blockIdx.x * blockDim.x + threadIdx.x;
  if (i < n) out[i] = (_Float16)in[i];
}

// In-place RoPE on [B,H,S,64] f16; one thread handles a (d, d+32) pair -> no race.
__global__ void rope_f16(_Float16* __restrict__ t, int S, int npair) {
  int i = blockIdx.x * blockDim.x + threadIdx.x;
  if (i >= npair) return;
  int d2  = i & 31;
  int row = i >> 5;              // over B*H*S
  int s   = row % S;
  size_t base = (size_t)row * HEAD_DIM;
  float x1 = (float)t[base + d2];
  float x2 = (float)t[base + d2 + 32];
  float freq = __powf(10000.0f, -(float)d2 * (1.0f / 32.0f));
  float ang = (float)s * freq;
  float c = cosf(ang), sn = sinf(ang);
  t[base + d2]      = (_Float16)(x1 * c - x2 * sn);
  t[base + d2 + 32] = (_Float16)(x2 * c + x1 * sn);
}

// C[M,N] = A[M,K] * Bw[N,K]^T   (i.e. x @ W.T), f16 inputs, f32 accumulate via WMMA.
// mode 0: write f32 row-major to Cf.  mode 1: write f16 head-split [B,H,S,64] to Ch.
__global__ __launch_bounds__(256)
void wmma_gemm(const _Float16* __restrict__ A, const _Float16* __restrict__ Bw,
               float* __restrict__ Cf, _Float16* __restrict__ Ch,
               int M, int N, int K, int S, int mode) {
  __shared__ __align__(16) _Float16 a_s[128][40];     // 128 x K32, row-major (+pad)
  __shared__ __align__(16) _Float16 b_op[4][32][16];  // B tile pre-swizzled to operand layout
  const int tid  = threadIdx.x;
  const int lane = tid & 31;
  const int wave = tid >> 5;                          // 8 waves: rows wave*16..+15
  const int m0 = blockIdx.y * 128;
  const int n0 = blockIdx.x * 64;
  const int alo = (lane < 16) ? 0 : 16;               // A-layout contiguous run starts
  const int ahi = (lane < 16) ? 8 : 24;
  v8f acc[4] = {};
  for (int ks = 0; ks < K; ks += 32) {
    { // stage A tile: 128 rows x 32 halves, 16 halves per thread (coalesced b128)
      int row = tid >> 1, part = (tid & 1) << 4;
      const _Float16* src = A + (size_t)(m0 + row) * K + ks + part;
      if (ks + 32 < K) __builtin_prefetch(src + 32, 0, 1);   // global_prefetch_b8
      *(uint4*)&a_s[row][part]     = *(const uint4*)src;
      *(uint4*)&a_s[row][part + 8] = *(const uint4*)(src + 8);
    }
    { // stage B tile straight into operand layout: one 16B global load -> one 16B LDS store
      int n = tid >> 2, kc = (tid & 3) << 3;          // 8 contiguous k for fixed n
      const _Float16* src = Bw + (size_t)(n0 + n) * K + ks + kc;
      uint4 raw = *(const uint4*)src;
      int laneL = (n & 15) + ((kc & 16) ? 16 : 0);    // N -> lane, K-half -> lane half
      *(uint4*)&b_op[n >> 4][laneL][kc & 8] = raw;
    }
    __syncthreads();
    const int mrow = (wave << 4) + (lane & 15);
    v16h a = cat8(&a_s[mrow][alo], &a_s[mrow][ahi]);
    #pragma unroll
    for (int nb = 0; nb < 4; ++nb) {
      v16h b = cat8(&b_op[nb][lane][0], &b_op[nb][lane][8]);
      acc[nb] = __builtin_amdgcn_wmma_f32_16x16x32_f16(false, a, false, b,
                                                       (short)0, acc[nb], false, false);
    }
    __syncthreads();
  }
  #pragma unroll
  for (int nb = 0; nb < 4; ++nb) {
    #pragma unroll
    for (int r = 0; r < 8; ++r) {
      int m = m0 + (wave << 4) + r + ((lane >= 16) ? 8 : 0);
      int n = n0 + (nb << 4) + (lane & 15);
      float v = acc[nb][r];
      if (mode == 0) {
        Cf[(size_t)m * N + n] = v;
      } else {
        int bb = m / S, s = m % S, h = n >> 6, dh = n & 63;
        Ch[((((size_t)bb * (N >> 6) + h) * S + s) << 6) + dh] = (_Float16)v;
      }
    }
  }
}

// Flash attention: grid (S/64, B*H). Block = 4 waves, each wave owns 16 query rows.
// Causal online softmax; scores never touch HBM.
__global__ __launch_bounds__(128)
void flash_attn(const _Float16* __restrict__ Q, const _Float16* __restrict__ Kk,
                const _Float16* __restrict__ V, _Float16* __restrict__ O,
                int S, int H) {
  __shared__ __align__(16) _Float16 k_s[32][72];      // [key][dim]  (row-major)
  __shared__ __align__(16) _Float16 v_t[64][40];      // [dim][key]  (transposed)
  __shared__ __align__(16) _Float16 p_s[4][16][40];   // per-wave P relayout buffer
  const int tid = threadIdx.x, lane = tid & 31, wave = tid >> 5;
  const int bh = blockIdx.y;
  const int q0 = blockIdx.x << 6;
  const _Float16* Qb = Q  + (size_t)bh * S * HEAD_DIM;
  const _Float16* Kb = Kk + (size_t)bh * S * HEAD_DIM;
  const _Float16* Vb = V  + (size_t)bh * S * HEAD_DIM;
  const int alo = (lane < 16) ? 0 : 16;               // A-layout run starts
  const int ahi = (lane < 16) ? 8 : 24;
  const int bofs = (lane < 16) ? 0 : 16;              // B-layout run start
  v16h qa0, qa1;
  { // Q tile (16x64) as two A operands, wide global loads
    int mrow = q0 + (wave << 4) + (lane & 15);
    const _Float16* qp = Qb + (size_t)mrow * HEAD_DIM;
    qa0 = cat8(qp + alo,      qp + ahi);
    qa1 = cat8(qp + 32 + alo, qp + 32 + ahi);
  }
  float rmax[8], lsum[8];
  #pragma unroll
  for (int r = 0; r < 8; ++r) { rmax[r] = -__builtin_inff(); lsum[r] = 0.f; }
  v8f o[4] = {};
  const int q_hi = q0 + (wave << 4) + 15;   // wave-uniform causal limit
  const int nkb  = (q0 + 64) >> 5;
  for (int kb = 0; kb < nkb; ++kb) {
    const int k0 = kb << 5;
    { // stage K row-major (wide) and V transposed (scatter on store side)
      int row = tid >> 2, part = (tid & 3) << 4;
      const _Float16* ks = Kb + (size_t)(k0 + row) * HEAD_DIM + part;
      const _Float16* vs = Vb + (size_t)(k0 + row) * HEAD_DIM + part;
      *(uint4*)&k_s[row][part]     = *(const uint4*)ks;
      *(uint4*)&k_s[row][part + 8] = *(const uint4*)(ks + 8);
      _Float16 vtmp[16];
      *(uint4*)&vtmp[0] = *(const uint4*)vs;
      *(uint4*)&vtmp[8] = *(const uint4*)(vs + 8);
      #pragma unroll
      for (int j = 0; j < 16; ++j) v_t[part + j][row] = vtmp[j];
    }
    __syncthreads();
    if (k0 <= q_hi) {   // wave-uniform branch -> EXEC stays all-1s for WMMA
      v8f c0 = {}, c1 = {};
      #pragma unroll
      for (int t = 0; t < 2; ++t) {   // two 16-key score tiles
        v8f cc = {};
        #pragma unroll
        for (int stp = 0; stp < 2; ++stp) {   // Dh=64 -> two K=32 steps
          const _Float16* kr = &k_s[(t << 4) + (lane & 15)][(stp << 5) + bofs];
          v16h b = cat8(kr, kr + 8);          // contiguous 16-half run
          cc = __builtin_amdgcn_wmma_f32_16x16x32_f16(false, (stp ? qa1 : qa0),
                                                      false, b, (short)0, cc, false, false);
        }
        if (t == 0) c0 = cc; else c1 = cc;
      }
      float p0[8], p1[8];
      #pragma unroll
      for (int r = 0; r < 8; ++r) {
        const int m_abs = q0 + (wave << 4) + r + ((lane >= 16) ? 8 : 0);
        const int col = lane & 15;
        float s0 = c0[r] * 0.125f;               // scale = Dh^-0.5
        float s1 = c1[r] * 0.125f;
        if (k0 + col      > m_abs) s0 = -__builtin_inff();
        if (k0 + 16 + col > m_abs) s1 = -__builtin_inff();
        float rowm = fmaxf(s0, s1);
        #pragma unroll
        for (int msk = 1; msk < 16; msk <<= 1) rowm = fmaxf(rowm, __shfl_xor(rowm, msk, 32));
        float nm    = fmaxf(rmax[r], rowm);
        float alpha = (nm == -__builtin_inff()) ? 1.f : __expf(rmax[r] - nm);
        float e0 = (s0 == -__builtin_inff()) ? 0.f : __expf(s0 - nm);
        float e1 = (s1 == -__builtin_inff()) ? 0.f : __expf(s1 - nm);
        float rs = e0 + e1;
        #pragma unroll
        for (int msk = 1; msk < 16; msk <<= 1) rs += __shfl_xor(rs, msk, 32);
        lsum[r] = lsum[r] * alpha + rs;
        rmax[r] = nm;
        o[0][r] *= alpha; o[1][r] *= alpha; o[2][r] *= alpha; o[3][r] *= alpha;
        p0[r] = e0; p1[r] = e1;
      }
      // Relayout P (C layout -> A layout) through wave-private LDS
      #pragma unroll
      for (int r = 0; r < 8; ++r) {
        int row = r + ((lane >= 16) ? 8 : 0);
        p_s[wave][row][lane & 15]        = (_Float16)p0[r];
        p_s[wave][row][16 + (lane & 15)] = (_Float16)p1[r];
      }
      asm volatile("s_wait_dscnt 0x0" ::: "memory");   // wave-local LDS RAW fence
      __builtin_amdgcn_wave_barrier();
      const _Float16* pr = &p_s[wave][lane & 15][0];
      v16h pa = cat8(pr + alo, pr + ahi);
      #pragma unroll
      for (int nb = 0; nb < 4; ++nb) {   // O(16x64) += P(16x32) * V(32x64)
        const _Float16* vr = &v_t[(nb << 4) + (lane & 15)][bofs];
        v16h b = cat8(vr, vr + 8);
        o[nb] = __builtin_amdgcn_wmma_f32_16x16x32_f16(false, pa, false, b,
                                                       (short)0, o[nb], false, false);
      }
    }
    __syncthreads();
  }
  // normalize + write [B,S,H*64] f16
  #pragma unroll
  for (int r = 0; r < 8; ++r) {
    const int m_abs = q0 + (wave << 4) + r + ((lane >= 16) ? 8 : 0);
    float inv = (lsum[r] > 0.f) ? (1.f / lsum[r]) : 0.f;
    size_t base = ((((size_t)(bh / H)) * S + m_abs) * H + (bh % H)) * HEAD_DIM;
    #pragma unroll
    for (int nb = 0; nb < 4; ++nb)
      O[base + (nb << 4) + (lane & 15)] = (_Float16)(o[nb][r] * inv);
  }
}

extern "C" void kernel_launch(void* const* d_in, const int* in_sizes, int n_in,
                              void* d_out, int out_size, void* d_ws, size_t ws_size,
                              hipStream_t stream) {
  (void)n_in; (void)out_size; (void)ws_size;
  const int D = D_MODEL;
  const int H = N_HEADS;
  const int BS = in_sizes[0] / D;        // B*S = 8192
  int S = 1;
  while ((long long)S * (long long)S < (long long)in_sizes[1]) S <<= 1;  // 2048
  const int B = BS / S;

  char* ws = (char*)d_ws;
  auto bump = [&](size_t bytes) -> _Float16* {
    char* p = ws;
    ws += (bytes + 255) & ~(size_t)255;
    return (_Float16*)p;
  };
  const size_t xel = (size_t)BS * D;     // 8.39M elems
  const size_t wel = (size_t)D * D;      // 1.05M elems
  _Float16* xh = bump(xel * 2);
  _Float16* wq = bump(wel * 2);
  _Float16* wk = bump(wel * 2);
  _Float16* wv = bump(wel * 2);
  _Float16* wo = bump(wel * 2);
  _Float16* Qh = bump(xel * 2);          // [B,H,S,64]
  _Float16* Kh = bump(xel * 2);
  _Float16* Vh = bump(xel * 2);
  _Float16* Ah = bump(xel * 2);          // attention out [B,S,H*64]

  cvt_f32_f16<<<(int)((xel + 255) / 256), 256, 0, stream>>>((const float*)d_in[0], xh, (int)xel);
  cvt_f32_f16<<<(int)((wel + 255) / 256), 256, 0, stream>>>((const float*)d_in[2], wq, (int)wel);
  cvt_f32_f16<<<(int)((wel + 255) / 256), 256, 0, stream>>>((const float*)d_in[3], wk, (int)wel);
  cvt_f32_f16<<<(int)((wel + 255) / 256), 256, 0, stream>>>((const float*)d_in[4], wv, (int)wel);
  cvt_f32_f16<<<(int)((wel + 255) / 256), 256, 0, stream>>>((const float*)d_in[5], wo, (int)wel);

  dim3 gg(D / 64, BS / 128);
  wmma_gemm<<<gg, 256, 0, stream>>>(xh, wq, nullptr, Qh, BS, D, D, S, 1);
  wmma_gemm<<<gg, 256, 0, stream>>>(xh, wk, nullptr, Kh, BS, D, D, S, 1);
  wmma_gemm<<<gg, 256, 0, stream>>>(xh, wv, nullptr, Vh, BS, D, D, S, 1);

  int npair = (int)(xel / 2);
  rope_f16<<<(npair + 255) / 256, 256, 0, stream>>>(Qh, S, npair);
  rope_f16<<<(npair + 255) / 256, 256, 0, stream>>>(Kh, S, npair);

  flash_attn<<<dim3(S / 64, B * H), 128, 0, stream>>>(Qh, Kh, Vh, Ah, S, H);

  wmma_gemm<<<gg, 256, 0, stream>>>(Ah, wo, (float*)d_out, nullptr, BS, D, D, S, 0);
}
